// GeodesicKernelUpsample_66305705116311
// MI455X (gfx1250) — matrix operand order repaired and checked by reading, not compiled, not run on hardware
//
#include <hip/hip_runtime.h>
#include <hip/hip_bf16.h>

// ---- problem constants (match reference) ----
constexpr int   B_    = 4;
constexpr int   N_IN  = 10242;
constexpr int   N_OUT = 40962;
constexpr int   K_    = 7;
constexpr int   C_    = 128;
constexpr float SIGMA = 0.4f;

// ---- tiling ----
constexpr int TILE      = 16;    // output rows per wave (WMMA M)
constexpr int SLOTS     = 128;   // 16 rows * 8 K-slots (7 weights + 1 pad)
constexpr int PITCH     = 136;   // halfs per LDS channel row (128 + 8 pad)
constexpr int OUT_PITCH = 132;   // floats per LDS out row (128 + 4 pad):
                                 // 8*132 mod 64 = 32 -> lane halves on disjoint banks

typedef __attribute__((ext_vector_type(16))) _Float16 v16h;
typedef __attribute__((ext_vector_type(8)))  _Float16 v8h;
typedef __attribute__((ext_vector_type(8)))  float    v8f;
typedef __attribute__((ext_vector_type(4)))  float    v4f;

// One wave (32 lanes) per workgroup. Handles 16 output points, all 4 batches,
// all 128 channels. Weighted 7-tap combine expressed as block-diagonal
// 16x128 (f16) x 128x16 (f16) WMMA chain with f32 accumulation.
__global__ __launch_bounds__(32)
void geo_upsample_wmma(const float* __restrict__ x,      // [B, N_IN, C]
                       const float* __restrict__ delta,  // [N_OUT, K]
                       const float* __restrict__ mask,   // [N_OUT, K]
                       const int*   __restrict__ idx,    // [N_OUT, K]
                       float*       __restrict__ out)    // [B, N_OUT, C]
{
    // LDS: gathered features, TRANSPOSED: [channel 0..127][slot 0..127] f16
    __shared__ __align__(16) _Float16 Bt[C_ * PITCH];
    // LDS: output staging tile [row 0..15][channel 0..127] f32 (padded pitch)
    __shared__ __align__(16) float OutS[TILE * OUT_PITCH];

    const int lane = threadIdx.x;        // 0..31
    const int m    = lane & 15;          // row within tile (WMMA M for A)
    const int h    = lane >> 4;          // lane half
    const int n0   = blockIdx.x * TILE;

    // ---------- zero the 16 pad slots (k==7) once; never rewritten ----------
#pragma unroll
    for (int t = 0; t < TILE; ++t) {
        const int c = lane * 4;
        const int s = 8 * t + 7;
        Bt[(c + 0) * PITCH + s] = (_Float16)0.0f;
        Bt[(c + 1) * PITCH + s] = (_Float16)0.0f;
        Bt[(c + 2) * PITCH + s] = (_Float16)0.0f;
        Bt[(c + 3) * PITCH + s] = (_Float16)0.0f;
    }

    // ---------- per-tile weights (f32 -> normalized -> f16) ----------
    int nr = n0 + m;
    if (nr >= N_OUT) nr = N_OUT - 1;

    const float inv2s2 = 1.0f / (2.0f * SIGMA * SIGMA);
    float wf[8];
    float wsum = 0.0f;
#pragma unroll
    for (int k = 0; k < K_; ++k) {
        float d  = delta[nr * K_ + k];
        float wk = __expf(-(d * d) * inv2s2) * mask[nr * K_ + k];
        wf[k] = wk;
        wsum += wk;
    }
    wf[7] = 0.0f;                                   // pad slot
    const float inv = 1.0f / fmaxf(wsum, 1e-8f);
    _Float16 wh[8];
#pragma unroll
    for (int k = 0; k < 8; ++k) wh[k] = (_Float16)(wf[k] * inv);

    // ---------- build the 4 block-diagonal A chunks (16x32 f16 each) ----------
    // Row m's weights occupy K-slots [8m, 8m+8). Chunk j holds K [32j,32j+32):
    // active rows m in [4j,4j+4), t=m&3; active iff (t&1)==h; elem half = t>>1.
    v16h A[4];
#pragma unroll
    for (int j = 0; j < 4; ++j) {
        v16h a = {};
        const bool active = ((m >> 2) == j) && ((m & 1) == h);
        const bool hiHalf = (m & 2) != 0;
#pragma unroll
        for (int k = 0; k < 8; ++k) {
            a[k]     = (active && !hiHalf) ? wh[k] : (_Float16)0.0f;
            a[k + 8] = (active &&  hiHalf) ? wh[k] : (_Float16)0.0f;
        }
        A[j] = a;
    }

    // ---------- per-batch: stage gathers to LDS, WMMA, coalesced store ----------
    for (int b = 0; b < B_; ++b) {
        // stage: slot s = 8*t + k, tap k in [0,7)
        const float* __restrict__ xb = x + (size_t)b * N_IN * C_;
        for (int t = 0; t < TILE; ++t) {
            int n = n0 + t;
            if (n >= N_OUT) n = N_OUT - 1;
#pragma unroll
            for (int k = 0; k < K_; ++k) {
                const int r = idx[n * K_ + k];           // wave-uniform -> s_load
                const v4f* p = (const v4f*)(xb + (size_t)r * C_);
                const v4f v = p[lane];                   // coalesced b128 gather
                const int c = lane * 4;                  // transpose into LDS
                const int s = 8 * t + k;
                Bt[(c + 0) * PITCH + s] = (_Float16)v[0];
                Bt[(c + 1) * PITCH + s] = (_Float16)v[1];
                Bt[(c + 2) * PITCH + s] = (_Float16)v[2];
                Bt[(c + 3) * PITCH + s] = (_Float16)v[3];
            }
        }
        // single-wave workgroup: DS ops are in-order within the wave -> no barrier.

        // compute: 8 channel blocks x 4 K-chunks of v_wmma_f32_16x16x32_f16
#pragma unroll
        for (int cb = 0; cb < 8; ++cb) {
            v8f acc = {};
#pragma unroll
            for (int j = 0; j < 4; ++j) {
                // B-operand (32x16 f16): lane = column N=(lane&15) = channel,
                // elements 0..7 = K = 8h + e, 8..15 = K = 16 + 8h + (e-8).
                const _Float16* bp =
                    &Bt[(cb * 16 + (lane & 15)) * PITCH + 32 * j + 8 * h];
                v8h lo = *(const v8h*)(bp);          // ds_load_b128
                v8h hi = *(const v8h*)(bp + 16);     // ds_load_b128
                v16h Bb = __builtin_shufflevector(lo, hi,
                    0, 1, 2, 3, 4, 5, 6, 7, 8, 9, 10, 11, 12, 13, 14, 15);
                acc = __builtin_amdgcn_wmma_f32_16x16x32_f16(
                    /*neg_a=*/false, A[j], /*neg_b=*/false, Bb,
                    /*c_mod=*/(short)0, acc,
                    /*reuse_a=*/false, /*reuse_b=*/false);
            }
            // D layout: lane = column N=(lane&15)=channel, VGPR v -> row 8h+v.
            // Scatter into LDS out tile (disjoint banks for the two lane halves).
            const int c = cb * 16 + (lane & 15);
#pragma unroll
            for (int v = 0; v < 8; ++v) {
                OutS[(8 * h + v) * OUT_PITCH + c] = acc[v];
            }
        }

        // coalesced write-out: 16 rows x 512B, uniform guard (scalar branch),
        // non-temporal so the 84MB output stream doesn't evict x from L2.
        float* __restrict__ ob = out + (size_t)b * N_OUT * C_;
#pragma unroll
        for (int r = 0; r < TILE; ++r) {
            const int n = n0 + r;                        // wave-uniform guard
            if (n < N_OUT) {
                const v4f val = *(const v4f*)&OutS[r * OUT_PITCH + lane * 4];
                __builtin_nontemporal_store(
                    val, (v4f*)(ob + (size_t)n * C_ + lane * 4));
            }
        }
    }
}

extern "C" void kernel_launch(void* const* d_in, const int* in_sizes, int n_in,
                              void* d_out, int out_size, void* d_ws, size_t ws_size,
                              hipStream_t stream) {
    const float* x     = (const float*)d_in[0];   // [B, N_IN, C] f32
    const float* delta = (const float*)d_in[1];   // [N_OUT, K]   f32
    const float* mask  = (const float*)d_in[2];   // [N_OUT, K]   f32
    const int*   idx   = (const int*)  d_in[3];   // [N_OUT, K]   i32
    float*       out   = (float*)d_out;           // [B, N_OUT, C] f32

    const int tiles = (N_OUT + TILE - 1) / TILE;  // 2561
    hipLaunchKernelGGL(geo_upsample_wmma, dim3(tiles), dim3(32), 0, stream,
                       x, delta, mask, idx, out);
}
